// NonLocalBlock_16320875725492
// MI455X (gfx1250) — compile-verified, matrix operand order checked by reading, not run on hardware
//
#include <hip/hip_runtime.h>
#include <hip/hip_bf16.h>

#define B_   4
#define C_   256
#define CI_  128
#define NPOS 4096   // H*W = 64*64
#define EPS_ 1e-5f

typedef _Float16 half_t;
typedef __attribute__((ext_vector_type(16))) _Float16 v16h;
typedef __attribute__((ext_vector_type(8)))  _Float16 v8h;
typedef __attribute__((ext_vector_type(8)))  float    v8f;
typedef __attribute__((ext_vector_type(4)))  unsigned int v4u;
typedef __attribute__((ext_vector_type(8)))  int          v8i;
typedef __attribute__((ext_vector_type(4)))  int          v4i;

// ---- lane-xor within 16-lane rows: v_permlane16_b32 (VALU, no LDS round-trip) ----
__device__ __forceinline__ float lanexor16(float v, unsigned selLo, unsigned selHi, int m) {
#if __has_builtin(__builtin_amdgcn_permlane16)
    unsigned u = __builtin_bit_cast(unsigned, v);
    u = __builtin_amdgcn_permlane16(u, u, selLo, selHi, false, false);
    return __builtin_bit_cast(float, u);
#else
    return __shfl_xor(v, m, 32);
#endif
}
#define XMAX(t) do { \
    t = fmaxf(t, lanexor16(t, 0x67452301u, 0xEFCDAB89u, 1)); \
    t = fmaxf(t, lanexor16(t, 0x54761032u, 0xDCFE98BAu, 2)); \
    t = fmaxf(t, lanexor16(t, 0x32107654u, 0xBA98FEDCu, 4)); \
    t = fmaxf(t, lanexor16(t, 0xFEDCBA98u, 0x76543210u, 8)); \
} while (0)
#define XSUM(t) do { \
    t += lanexor16(t, 0x67452301u, 0xEFCDAB89u, 1); \
    t += lanexor16(t, 0x54761032u, 0xDCFE98BAu, 2); \
    t += lanexor16(t, 0x32107654u, 0xBA98FEDCu, 4); \
    t += lanexor16(t, 0xFEDCBA98u, 0x76543210u, 8); \
} while (0)

// Low 32 bits of a flat LDS address = LDS byte offset (aperture is the hi dword).
__device__ __forceinline__ unsigned int lds_offset(const void* p) {
    return (unsigned int)(uintptr_t)p;
}

// Issue a 2D TENSOR_LOAD_TO_LDS: tile (tile_d1 rows x tile_d0 elems) of 2-byte
// elements, row stride stride_d0 elems in global, optional LDS row padding.
__device__ __forceinline__ void tdm_load_2d(
    unsigned int lds_addr, const void* gaddr,
    unsigned int tensor_d0, unsigned int tensor_d1, unsigned long long stride_d0,
    unsigned int tile_d0, unsigned int tile_d1,
    unsigned int pad_interval, unsigned int pad_amount)
{
    unsigned long long ga = (unsigned long long)(uintptr_t)gaddr;
    v4u g0;
    g0[0] = 1u;                                   // count=1, user descriptor
    g0[1] = lds_addr;                             // LDS byte address
    g0[2] = (unsigned int)ga;                     // global_addr[31:0]
    g0[3] = (unsigned int)((ga >> 32) & 0x1FFFFFFu) | (2u << 30);  // addr[56:32] | type=2
    v8i g1;
    g1[0] = (int)((1u << 16)                      // data_size = 1 -> 2 bytes
                | (1u << 20)                      // pad_enable
                | (pad_interval << 22)
                | (pad_amount << 25));
    g1[1] = (int)((tensor_d0 & 0xFFFFu) << 16);                               // dim0[15:0]
    g1[2] = (int)(((tensor_d0 >> 16) & 0xFFFFu) | ((tensor_d1 & 0xFFFFu) << 16));
    g1[3] = (int)(((tensor_d1 >> 16) & 0xFFFFu) | ((tile_d0 & 0xFFFFu) << 16));
    g1[4] = (int)(tile_d1 & 0xFFFFu);             // tile_dim1, tile_dim2=0
    g1[5] = (int)(unsigned int)(stride_d0 & 0xFFFFFFFFu);
    g1[6] = (int)(unsigned int)((stride_d0 >> 32) & 0xFFFFu);
    g1[7] = 0;
    v4i z4 = {0, 0, 0, 0};
#if defined(__clang_major__) && (__clang_major__ >= 23)
    v8i z8 = {0, 0, 0, 0, 0, 0, 0, 0};
    __builtin_amdgcn_tensor_load_to_lds(g0, g1, z4, z4, z8, 0);
#else
    __builtin_amdgcn_tensor_load_to_lds(g0, g1, z4, z4, 0);
#endif
}

// Load a 32-wide (K-dim) f16 fragment row for the CDNA5 16-bit A/B layouts:
// lane-half selects K subsets {0-7,16-23} vs {8-15,24-31}.
__device__ __forceinline__ v16h load_frag32(const half_t* __restrict__ row, int hiOff) {
    v8h a = *(const v8h*)(row + hiOff);
    v8h b = *(const v8h*)(row + 16 + hiOff);
    v16h r;
#pragma unroll
    for (int i = 0; i < 8; ++i) { r[i] = a[i]; r[8 + i] = b[i]; }
    return r;
}

// ---------------- Kernel 1: fused QKV 1x1-conv projection ----------------
// x:(B,C,N) f32 -> qT,kT:(B,N,CI) f16 (row-major channels), v:(B,CI,N) f16
__global__ __launch_bounds__(256) void qkv_proj_kernel(
    const float* __restrict__ x,
    const float* __restrict__ Wq, const float* __restrict__ bq,
    const float* __restrict__ Wk, const float* __restrict__ bk,
    const float* __restrict__ Wv, const float* __restrict__ bv,
    half_t* __restrict__ qT, half_t* __restrict__ kT, half_t* __restrict__ vv)
{
    __shared__ float xs[C_ * 64];      // [c][n], 64 KB
    const int b  = blockIdx.y;
    const int n0 = blockIdx.x * 64;
    const int tid = threadIdx.x;
#pragma unroll
    for (int i = 0; i < (C_ * 64) / 256; ++i) {
        int idx = tid + i * 256;
        int c = idx >> 6, n = idx & 63;
        xs[idx] = x[((size_t)(b * C_ + c)) * NPOS + n0 + n];
    }
    __syncthreads();
    const int n  = tid & 63;
    const int t4 = tid >> 6;
    for (int j = 0; j < 96; ++j) {
        int ocAll = j * 4 + t4;          // 0..383 => 3 projections x 128 ch
        int p  = ocAll >> 7;
        int ci = ocAll & 127;
        const float* wrow; float acc;
        if (p == 0)      { wrow = Wq + ci * C_; acc = bq[ci]; }
        else if (p == 1) { wrow = Wk + ci * C_; acc = bk[ci]; }
        else             { wrow = Wv + ci * C_; acc = bv[ci]; }
#pragma unroll 8
        for (int c = 0; c < C_; ++c) acc += wrow[c] * xs[c * 64 + n];
        if (p == 0)      qT[((size_t)(b * NPOS + n0 + n)) * CI_ + ci] = (half_t)acc;
        else if (p == 1) kT[((size_t)(b * NPOS + n0 + n)) * CI_ + ci] = (half_t)acc;
        else             vv[((size_t)(b * CI_ + ci)) * NPOS + n0 + n] = (half_t)acc;
    }
}

// ---------------- Kernel 2: flash attention, TDM double-buffered ----------------
__global__ __launch_bounds__(256) void attention_kernel(
    const half_t* __restrict__ qT, const half_t* __restrict__ kT,
    const half_t* __restrict__ vv, float* __restrict__ Oattn)
{
    // Padded LDS rows (stride not multiple of 64 banks); TDM pad feature writes
    // the same layout. 17.4K + 20.5K + 10.2K = 48.1 KB static LDS.
    __shared__ __align__(16) half_t kbuf[2][32][CI_ + 8];   // [buf][key][ch]
    __shared__ __align__(16) half_t vbuf[2][CI_][32 + 8];   // [buf][ch][key]
    __shared__ __align__(16) half_t pbuf[8][16][32 + 8];    // per-wave P transpose

    const int b     = blockIdx.y;
    const int wave  = threadIdx.x >> 5;
    const int lane  = threadIdx.x & 31;
    const int hi    = (lane >> 4) & 1;
    const int hiOff = hi * 8;
    const int col   = lane & 15;
    const int q0    = blockIdx.x * 128 + wave * 16;   // 16 query rows per wave
    const size_t bOff = (size_t)b * NPOS;

    // Q fragments: A-matrix 16x32 per K-chunk, 4 chunks cover CI=128
    v16h qf[4];
    {
        const half_t* qrow = qT + (bOff + q0 + col) * CI_;
#pragma unroll
        for (int kc = 0; kc < 4; ++kc) qf[kc] = load_frag32(qrow + kc * 32, hiOff);
    }

    v8f acc[8];
#pragma unroll
    for (int ch = 0; ch < 8; ++ch)
#pragma unroll
        for (int j = 0; j < 8; ++j) acc[ch][j] = 0.0f;
    float mrun[8], lrun[8];
#pragma unroll
    for (int j = 0; j < 8; ++j) { mrun[j] = -3.0e38f; lrun[j] = 0.0f; }

    // Prologue: DMA first K/V tile into buffer 0 (wave 0 issues; EXEC-independent).
    if (wave == 0) {
        tdm_load_2d(lds_offset(&kbuf[0][0][0]), kT + bOff * CI_,
                    CI_, 32, CI_, CI_, 32, /*pad: 64 dw*/ 5, /*4 dw*/ 3);
        tdm_load_2d(lds_offset(&vbuf[0][0][0]), vv + (size_t)b * CI_ * NPOS,
                    32, CI_, NPOS, 32, CI_, /*pad: 16 dw*/ 3, /*4 dw*/ 3);
    }

    for (int kt = 0; kt < NPOS; kt += 32) {
        const int buf = (kt >> 5) & 1;
        if (wave == 0) {
            if (kt + 32 < NPOS) {   // prefetch next tile into other buffer
                const int nb = buf ^ 1;
                tdm_load_2d(lds_offset(&kbuf[nb][0][0]),
                            kT + (bOff + kt + 32) * CI_,
                            CI_, 32, CI_, CI_, 32, 5, 3);
                tdm_load_2d(lds_offset(&vbuf[nb][0][0]),
                            vv + (size_t)b * CI_ * NPOS + kt + 32,
                            32, CI_, NPOS, 32, CI_, 3, 3);
                __builtin_amdgcn_s_wait_tensorcnt(2);  // current tile landed
            } else {
                __builtin_amdgcn_s_wait_tensorcnt(0);
            }
        }
        __syncthreads();

        // ---- S = Q @ K^T (two 16-key half tiles) from LDS ----
        v8f S0, S1;
#pragma unroll
        for (int j = 0; j < 8; ++j) { S0[j] = 0.0f; S1[j] = 0.0f; }
        const half_t* krow0 = &kbuf[buf][col][0];
        const half_t* krow1 = &kbuf[buf][16 + col][0];
#pragma unroll
        for (int kc = 0; kc < 4; ++kc) {
            v16h kf0 = load_frag32(krow0 + kc * 32, hiOff);
            S0 = __builtin_amdgcn_wmma_f32_16x16x32_f16(false, qf[kc], false, kf0,
                                                        (short)0, S0, false, false);
            v16h kf1 = load_frag32(krow1 + kc * 32, hiOff);
            S1 = __builtin_amdgcn_wmma_f32_16x16x32_f16(false, qf[kc], false, kf1,
                                                        (short)0, S1, false, false);
        }

        // ---- streaming softmax; rows live in one 16-lane half -> permlane16 ----
        float scal[8];
#pragma unroll
        for (int j = 0; j < 8; ++j) {
            float t = fmaxf(S0[j], S1[j]);
            XMAX(t);
            float mnew = fmaxf(mrun[j], t);
            scal[j] = __expf(mrun[j] - mnew);
            float p0 = __expf(S0[j] - mnew);
            float p1 = __expf(S1[j] - mnew);
            pbuf[wave][hi * 8 + j][col]      = (half_t)p0;   // C-layout -> LDS
            pbuf[wave][hi * 8 + j][col + 16] = (half_t)p1;
            float rs = p0 + p1;
            XSUM(rs);
            lrun[j] = lrun[j] * scal[j] + rs;
            mrun[j] = mnew;
        }
#pragma unroll
        for (int ch = 0; ch < 8; ++ch)
#pragma unroll
            for (int j = 0; j < 8; ++j) acc[ch][j] *= scal[j];

        // ---- reload P as A-fragment (per-wave transpose; DS ops in-order) ----
        asm volatile("s_wait_dscnt 0" ::: "memory");
        v16h pf;
#pragma unroll
        for (int i = 0; i < 8; ++i) {
            pf[i]     = pbuf[wave][col][hiOff + i];
            pf[8 + i] = pbuf[wave][col][16 + hiOff + i];
        }

        // ---- O += P @ V (8 channel chunks of 16) from LDS ----
#pragma unroll
        for (int ch = 0; ch < 8; ++ch) {
            const half_t* vrow = &vbuf[buf][ch * 16 + col][0];
            v16h vf = load_frag32(vrow, hiOff);
            acc[ch] = __builtin_amdgcn_wmma_f32_16x16x32_f16(false, pf, false, vf,
                                                             (short)0, acc[ch], false, false);
        }
        __syncthreads();   // tile consumed; TDM may overwrite this buffer next iter
    }

    // normalize by row sum and store (B,N,CI) f32
#pragma unroll
    for (int j = 0; j < 8; ++j) lrun[j] = 1.0f / lrun[j];
    const int rowbase = q0 + hi * 8;
#pragma unroll
    for (int ch = 0; ch < 8; ++ch)
#pragma unroll
        for (int j = 0; j < 8; ++j)
            Oattn[(bOff + rowbase + j) * CI_ + ch * 16 + col] = acc[ch][j] * lrun[j];
}

// ---------------- Kernel 3: zero BN statistics ----------------
__global__ void zero_stats_kernel(float* stats) {
    int i = blockIdx.x * blockDim.x + threadIdx.x;
    if (i < 2 * C_) stats[i] = 0.0f;
}

// ---------------- Kernel 4: output projection + BN partial sums ----------------
__global__ __launch_bounds__(256) void out_proj_kernel(
    const float* __restrict__ Oattn, const float* __restrict__ Wo,
    const float* __restrict__ bo, float* __restrict__ out2,
    float* __restrict__ stats)
{
    __shared__ float tile[64 * 129];   // [n][c] padded vs bank conflicts
    const int b  = blockIdx.y;
    const int n0 = blockIdx.x * 64;
    const int tid = threadIdx.x;
#pragma unroll
    for (int i = 0; i < (64 * CI_) / 256; ++i) {
        int idx = tid + i * 256;
        int n = idx >> 7, c = idx & 127;
        tile[n * 129 + c] = Oattn[((size_t)(b * NPOS + n0 + n)) * CI_ + c];
    }
    __syncthreads();
    const int n  = tid & 63;
    const int t4 = tid >> 6;
    float* sum   = stats;
    float* sumsq = stats + C_;
    for (int j = 0; j < 64; ++j) {
        int o = j * 4 + t4;              // uniform per wave
        float acc = bo[o];
        const float* wrow = Wo + o * CI_;
#pragma unroll 8
        for (int c = 0; c < CI_; ++c) acc += wrow[c] * tile[n * 129 + c];
        out2[((size_t)(b * C_ + o)) * NPOS + n0 + n] = acc;
        float s1 = acc, s2 = acc * acc;
        XSUM(s1); XSUM(s2);
        s1 += lanexor16(s1, 0, 0, 16) * 0.0f;  // no-op keep (halves combined below)
        // combine the two 16-lane halves with a single shuffle-xor-16
        s1 += __shfl_xor(s1, 16, 32);
        s2 += __shfl_xor(s2, 16, 32);
        if ((tid & 31) == 0) { atomicAdd(&sum[o], s1); atomicAdd(&sumsq[o], s2); }
    }
}

// ---------------- Kernel 5: BatchNorm (batch stats) + residual ----------------
__global__ __launch_bounds__(256) void bn_residual_kernel(
    const float* __restrict__ out2, const float* __restrict__ x,
    const float* __restrict__ gamma, const float* __restrict__ beta,
    const float* __restrict__ stats, float* __restrict__ out)
{
    size_t idx = (size_t)blockIdx.x * blockDim.x + threadIdx.x;
    const size_t total = (size_t)B_ * C_ * NPOS;
    if (idx >= total) return;
    int o = (int)((idx / NPOS) % C_);
    const float invN = 1.0f / (float)(B_ * NPOS);
    float mean = stats[o] * invN;
    float var  = stats[C_ + o] * invN - mean * mean;
    float r = rsqrtf(var + EPS_);
    out[idx] = (out2[idx] - mean) * r * gamma[o] + beta[o] + x[idx];
}

extern "C" void kernel_launch(void* const* d_in, const int* in_sizes, int n_in,
                              void* d_out, int out_size, void* d_ws, size_t ws_size,
                              hipStream_t stream) {
    const float* x     = (const float*)d_in[0];
    const float* Wq    = (const float*)d_in[1];
    const float* bq    = (const float*)d_in[2];
    const float* Wk    = (const float*)d_in[3];
    const float* bk    = (const float*)d_in[4];
    const float* Wv    = (const float*)d_in[5];
    const float* bv    = (const float*)d_in[6];
    const float* Wo    = (const float*)d_in[7];
    const float* bo    = (const float*)d_in[8];
    const float* gamma = (const float*)d_in[9];
    const float* beta  = (const float*)d_in[10];
    float* out = (float*)d_out;

    char* ws = (char*)d_ws;
    size_t off = 0;
    half_t* qT    = (half_t*)(ws + off); off += (size_t)B_ * NPOS * CI_ * sizeof(half_t);
    half_t* kT    = (half_t*)(ws + off); off += (size_t)B_ * NPOS * CI_ * sizeof(half_t);
    half_t* vv    = (half_t*)(ws + off); off += (size_t)B_ * NPOS * CI_ * sizeof(half_t);
    float*  Oattn = (float*)(ws + off);  off += (size_t)B_ * NPOS * CI_ * sizeof(float);
    float*  out2  = (float*)(ws + off);  off += (size_t)B_ * C_ * NPOS * sizeof(float);
    float*  stats = (float*)(ws + off);  off += 2 * C_ * sizeof(float);
    (void)ws_size; (void)in_sizes; (void)n_in; (void)out_size;

    zero_stats_kernel<<<2, 256, 0, stream>>>(stats);
    qkv_proj_kernel<<<dim3(NPOS / 64, B_), 256, 0, stream>>>(
        x, Wq, bq, Wk, bk, Wv, bv, qT, kT, vv);
    attention_kernel<<<dim3(NPOS / 128, B_), 256, 0, stream>>>(qT, kT, vv, Oattn);
    out_proj_kernel<<<dim3(NPOS / 64, B_), 256, 0, stream>>>(Oattn, Wo, bo, out2, stats);
    bn_residual_kernel<<<(int)(((size_t)B_ * C_ * NPOS + 255) / 256), 256, 0, stream>>>(
        out2, x, gamma, beta, stats, out);
}
